// AttentionBlock_74594991997625
// MI455X (gfx1250) — matrix operand order repaired and checked by reading, not compiled, not run on hardware
//
#include <hip/hip_runtime.h>
#include <hip/hip_bf16.h>

// ---------------------------------------------------------------------------
// Transformer encoder block on gfx1250 (CDNA5): all matmuls via
// v_wmma_f32_16x16x32_f16, flash-attention online softmax, f16 intermediates,
// global_prefetch_b8 streaming prefetch ~8 k-steps ahead.
// S=2048 B=4 DM=1024 H=16 DK=64 DFF=4096, rows M = S*B = 8192.
// ---------------------------------------------------------------------------

typedef _Float16 v16h __attribute__((ext_vector_type(16)));
typedef float    v8f  __attribute__((ext_vector_type(8)));

#define WMMA(a, b, c) __builtin_amdgcn_wmma_f32_16x16x32_f16( \
    false, (a), false, (b), (short)0, (c), false, false)

static constexpr int S_  = 2048;
static constexpr int B_  = 4;
static constexpr int DM_ = 1024;
static constexpr int H_  = 16;
static constexpr int DK_ = 64;
static constexpr int DFF_= 4096;
static constexpr int M_  = S_ * B_;   // 8192 rows
static constexpr int PF_ = 256;       // prefetch distance in K-elements (512B)

__device__ __forceinline__ void pf(const void* p) { __builtin_prefetch(p, 0, 1); }

// ---- A-operand fragment: 16x32 f16, rows striped per lane (ISA 7.12.2) ----
// lane 0-15: M=lane, K halves {0..7,16..23}; lane 16-31: same M, K +8.
__device__ __forceinline__ v16h load_fragA(const _Float16* src, int row0, int ld) {
  const int lane = threadIdx.x & 31;
  const int r    = row0 + (lane & 15);
  const int off  = (lane >> 4) * 8;
  const _Float16* p = src + (size_t)r * ld + off;
  union { v16h v; uint2 u[4]; } f;
  f.u[0] = *(const uint2*)(p + 0);    // K {0,1},{2,3}   (+half*8)
  f.u[1] = *(const uint2*)(p + 4);    // K {4,5},{6,7}
  f.u[2] = *(const uint2*)(p + 16);   // K {16..19}
  f.u[3] = *(const uint2*)(p + 20);   // K {20..23}
  return f.v;
}

// ---- B-operand fragment: 32x16 f16. Source is W[N,K] row-major; lane n
// holds column n of B = row n of W, K halves {0..7,16..23} (+8 for lanes 16-31). --
__device__ __forceinline__ v16h load_fragB(const _Float16* src, int row0, int ld) {
  const int lane = threadIdx.x & 31;
  const int r    = row0 + (lane & 15);
  const int off  = (lane >> 4) * 8;
  const _Float16* p = src + (size_t)r * ld + off;
  union { v16h v; uint2 u[4]; } f;
  f.u[0] = *(const uint2*)(p + 0);
  f.u[1] = *(const uint2*)(p + 4);
  f.u[2] = *(const uint2*)(p + 16);
  f.u[3] = *(const uint2*)(p + 20);
  return f.v;
}

// ---------------------------------------------------------------------------
// fp32 -> f16 bulk conversion, 4 elements/lane/iter (sizes are /4).
// ---------------------------------------------------------------------------
__global__ void cvt_f16_kernel(const float* __restrict__ in,
                               _Float16* __restrict__ out, int n) {
  int i = (blockIdx.x * blockDim.x + threadIdx.x) * 4;
  const int stride = gridDim.x * blockDim.x * 4;
  for (; i < n; i += stride) {
    const float4 v = *(const float4*)(in + i);
    union { _Float16 h[4]; uint2 u; } o;
    o.h[0] = (_Float16)v.x; o.h[1] = (_Float16)v.y;
    o.h[2] = (_Float16)v.z; o.h[3] = (_Float16)v.w;
    *(uint2*)(out + i) = o.u;
  }
}

// ---------------------------------------------------------------------------
// Generic GEMM: C[M,N] = A[M,K] * W[N,K]^T + bias.  MODE 0: f32 out.
// MODE 1: f16 relu out.  Block 256 thr = 8 waves, tile 128x256, wave 64x64.
// ---------------------------------------------------------------------------
template<int MODE>
__global__ __launch_bounds__(256)
void gemm_wmma_kernel(const _Float16* __restrict__ A,
                      const _Float16* __restrict__ W,
                      const float* __restrict__ bias,
                      float* __restrict__ outF,
                      _Float16* __restrict__ outH,
                      int M, int N, int K) {
  const int wid  = threadIdx.x >> 5;
  const int lane = threadIdx.x & 31;
  const int half = lane >> 4;
  const int nl   = lane & 15;
  const int m0 = blockIdx.y * 128 + (wid >> 2) * 64;
  const int n0 = blockIdx.x * 256 + (wid & 3) * 64;

  v8f acc[4][4] = {};
  for (int kk = 0; kk < K; kk += 32) {
    if (kk + PF_ < K) {
#pragma unroll
      for (int i = 0; i < 4; i++)
        pf(A + (size_t)(m0 + i * 16 + nl) * K + kk + PF_);
#pragma unroll
      for (int j = 0; j < 4; j++)
        pf(W + (size_t)(n0 + j * 16 + nl) * K + kk + PF_);
    }
    v16h a[4], b[4];
#pragma unroll
    for (int i = 0; i < 4; i++) a[i] = load_fragA(A + kk, m0 + i * 16, K);
#pragma unroll
    for (int j = 0; j < 4; j++) b[j] = load_fragB(W + kk, n0 + j * 16, K);
#pragma unroll
    for (int i = 0; i < 4; i++)
#pragma unroll
      for (int j = 0; j < 4; j++)
        acc[i][j] = WMMA(a[i], b[j], acc[i][j]);
  }
#pragma unroll
  for (int i = 0; i < 4; i++)
#pragma unroll
    for (int j = 0; j < 4; j++)
#pragma unroll
      for (int v = 0; v < 8; v++) {
        const int m = m0 + i * 16 + v + 8 * half;
        const int n = n0 + j * 16 + nl;
        float val = acc[i][j][v] + bias[n];
        if (MODE == 1) {
          outH[(size_t)m * N + n] = (_Float16)fmaxf(val, 0.0f);
        } else {
          outF[(size_t)m * N + n] = val;
        }
      }
}

// ---------------------------------------------------------------------------
// Fused QKV GEMM: N = 3*DM = 3072.  Scatter epilogue:
//   Q -> [b*H+h][s][dk]   K -> [b*H+h][s][dk]   V -> transposed [b*H+h][dv][s]
// ---------------------------------------------------------------------------
__global__ __launch_bounds__(256)
void qkv_gemm_kernel(const _Float16* __restrict__ A,
                     const _Float16* __restrict__ W,
                     const float* __restrict__ bq,
                     const float* __restrict__ bk,
                     const float* __restrict__ bv,
                     _Float16* __restrict__ Qo,
                     _Float16* __restrict__ Ko,
                     _Float16* __restrict__ Vt) {
  const int K = DM_;
  const int wid  = threadIdx.x >> 5;
  const int lane = threadIdx.x & 31;
  const int half = lane >> 4;
  const int nl   = lane & 15;
  const int m0 = blockIdx.y * 128 + (wid >> 2) * 64;
  const int n0 = blockIdx.x * 256 + (wid & 3) * 64;

  v8f acc[4][4] = {};
  for (int kk = 0; kk < K; kk += 32) {
    if (kk + PF_ < K) {
#pragma unroll
      for (int i = 0; i < 4; i++)
        pf(A + (size_t)(m0 + i * 16 + nl) * K + kk + PF_);
#pragma unroll
      for (int j = 0; j < 4; j++)
        pf(W + (size_t)(n0 + j * 16 + nl) * K + kk + PF_);
    }
    v16h a[4], b[4];
#pragma unroll
    for (int i = 0; i < 4; i++) a[i] = load_fragA(A + kk, m0 + i * 16, K);
#pragma unroll
    for (int j = 0; j < 4; j++) b[j] = load_fragB(W + kk, n0 + j * 16, K);
#pragma unroll
    for (int i = 0; i < 4; i++)
#pragma unroll
      for (int j = 0; j < 4; j++)
        acc[i][j] = WMMA(a[i], b[j], acc[i][j]);
  }
#pragma unroll
  for (int i = 0; i < 4; i++)
#pragma unroll
    for (int j = 0; j < 4; j++)
#pragma unroll
      for (int v = 0; v < 8; v++) {
        const int m = m0 + i * 16 + v + 8 * half;     // row = s*B + b
        const int n = n0 + j * 16 + nl;               // 0..3071
        const int s  = m >> 2;
        const int bb = m & 3;
        float val = acc[i][j][v];
        if (n < DM_) {
          val += bq[n];
          const int h = n >> 6, dk = n & 63;
          Qo[(((size_t)(bb * H_ + h)) * S_ + s) * DK_ + dk] = (_Float16)val;
        } else if (n < 2 * DM_) {
          const int f = n - DM_;
          val += bk[f];
          const int h = f >> 6, dk = f & 63;
          Ko[(((size_t)(bb * H_ + h)) * S_ + s) * DK_ + dk] = (_Float16)val;
        } else {
          const int f = n - 2 * DM_;
          val += bv[f];
          const int h = f >> 6, dv = f & 63;
          Vt[(((size_t)(bb * H_ + h)) * DK_ + dv) * S_ + s] = (_Float16)val;
        }
      }
}

// ---------------------------------------------------------------------------
// Flash attention. grid = (S/128, B*H). 8 waves/block, 16 query rows per wave.
// scores = Q K^T / 8, online softmax, ctx = P V.  P relayout C->A frag goes
// through a wave-private LDS tile (DS ops are in-order within a wave).
// ---------------------------------------------------------------------------
__global__ __launch_bounds__(256)
void attn_kernel(const _Float16* __restrict__ Q,
                 const _Float16* __restrict__ Kb,
                 const _Float16* __restrict__ Vt,
                 _Float16* __restrict__ Ctx) {
  const int bh   = blockIdx.y;
  const int b    = bh >> 4;
  const int h    = bh & 15;
  const int wid  = threadIdx.x >> 5;
  const int lane = threadIdx.x & 31;
  const int half = lane >> 4;
  const int nl   = lane & 15;
  const int m0   = blockIdx.x * 128 + wid * 16;

  const _Float16* q  = Q  + (size_t)bh * S_ * DK_;
  const _Float16* k  = Kb + (size_t)bh * S_ * DK_;
  const _Float16* vt = Vt + (size_t)bh * DK_ * S_;

  __shared__ _Float16 pbuf[8][16 * 32];
  _Float16* pw = pbuf[wid];

  const v16h qa0 = load_fragA(q,      m0, DK_);   // dk 0..31
  const v16h qa1 = load_fragA(q + 32, m0, DK_);   // dk 32..63

  float mrun[8], lrun[8];
#pragma unroll
  for (int v = 0; v < 8; v++) { mrun[v] = -__builtin_inff(); lrun[v] = 0.0f; }
  v8f cacc[4] = {};

  for (int kt = 0; kt < S_; kt += 32) {
    if (kt + PF_ < S_) {
      pf(k  + (size_t)(kt + PF_ + lane) * DK_);      // next K rows (128B each)
      pf(vt + (size_t)lane * S_ + kt + PF_);         // V^T rows dv=0..31
      pf(vt + (size_t)(lane + 32) * S_ + kt + PF_);  // V^T rows dv=32..63
    }
    v8f sc0 = {}, sc1 = {};
    sc0 = WMMA(qa0, load_fragB(k,      kt,      DK_), sc0);
    sc0 = WMMA(qa1, load_fragB(k + 32, kt,      DK_), sc0);
    sc1 = WMMA(qa0, load_fragB(k,      kt + 16, DK_), sc1);
    sc1 = WMMA(qa1, load_fragB(k + 32, kt + 16, DK_), sc1);

#pragma unroll
    for (int v = 0; v < 8; v++) {
      float s0 = sc0[v] * 0.125f;          // 1/sqrt(64)
      float s1 = sc1[v] * 0.125f;
      float t = fmaxf(s0, s1);             // row max over this key tile
      t = fmaxf(t, __shfl_xor(t, 1));
      t = fmaxf(t, __shfl_xor(t, 2));
      t = fmaxf(t, __shfl_xor(t, 4));
      t = fmaxf(t, __shfl_xor(t, 8));
      const float mn   = fmaxf(mrun[v], t);
      const float corr = __expf(mrun[v] - mn);
      const float p0   = __expf(s0 - mn);
      const float p1   = __expf(s1 - mn);
      float ps = p0 + p1;
      ps += __shfl_xor(ps, 1);
      ps += __shfl_xor(ps, 2);
      ps += __shfl_xor(ps, 4);
      ps += __shfl_xor(ps, 8);
      lrun[v] = lrun[v] * corr + ps;
      mrun[v] = mn;
      cacc[0][v] *= corr; cacc[1][v] *= corr;
      cacc[2][v] *= corr; cacc[3][v] *= corr;
      const int row = v + 8 * half;
      pw[row * 32 + nl]      = (_Float16)p0;
      pw[row * 32 + nl + 16] = (_Float16)p1;
    }
    asm volatile("" ::: "memory");          // keep ds_store before ds_load
    const v16h pa = load_fragA(pw, 0, 32);  // P as A-operand 16x32
    asm volatile("" ::: "memory");
#pragma unroll
    for (int j = 0; j < 4; j++)
      cacc[j] = WMMA(pa, load_fragB(vt + kt, j * 16, S_), cacc[j]);
  }

  float inv[8];
#pragma unroll
  for (int v = 0; v < 8; v++) inv[v] = 1.0f / lrun[v];
#pragma unroll
  for (int j = 0; j < 4; j++)
#pragma unroll
    for (int v = 0; v < 8; v++) {
      const int srow = m0 + v + 8 * half;
      const int col  = h * DK_ + j * 16 + nl;   // head-major concat
      Ctx[((size_t)srow * B_ + b) * DM_ + col] = (_Float16)(cacc[j][v] * inv[v]);
    }
}

// ---------------------------------------------------------------------------
// LayerNorm(y + res) over DM=1024 per row; optional f16 copy of output.
// ---------------------------------------------------------------------------
__global__ __launch_bounds__(256)
void ln_kernel(const float* __restrict__ y, const float* __restrict__ res,
               const float* __restrict__ g, const float* __restrict__ bb,
               float* __restrict__ out, _Float16* __restrict__ outh) {
  const int row = blockIdx.x;
  const float* py = y   + (size_t)row * DM_;
  const float* pr = res + (size_t)row * DM_;
  float vals[4], s = 0.0f, s2 = 0.0f;
#pragma unroll
  for (int i = 0; i < 4; i++) {
    const int c = threadIdx.x + i * 256;
    const float v = py[c] + pr[c];
    vals[i] = v; s += v; s2 += v * v;
  }
#pragma unroll
  for (int msk = 1; msk < 32; msk <<= 1) {
    s  += __shfl_xor(s,  msk);
    s2 += __shfl_xor(s2, msk);
  }
  __shared__ float sh[2][8];
  const int wid = threadIdx.x >> 5;
  if ((threadIdx.x & 31) == 0) { sh[0][wid] = s; sh[1][wid] = s2; }
  __syncthreads();
  s = 0.0f; s2 = 0.0f;
#pragma unroll
  for (int i = 0; i < 8; i++) { s += sh[0][i]; s2 += sh[1][i]; }
  const float mu   = s * (1.0f / DM_);
  const float var  = s2 * (1.0f / DM_) - mu * mu;
  const float rstd = rsqrtf(var + 1e-5f);
#pragma unroll
  for (int i = 0; i < 4; i++) {
    const int c = threadIdx.x + i * 256;
    const float o = (vals[i] - mu) * rstd * g[c] + bb[c];
    out[(size_t)row * DM_ + c] = o;
    if (outh) outh[(size_t)row * DM_ + c] = (_Float16)o;
  }
}

// ---------------------------------------------------------------------------
extern "C" void kernel_launch(void* const* d_in, const int* in_sizes, int n_in,
                              void* d_out, int out_size, void* d_ws, size_t ws_size,
                              hipStream_t stream) {
  (void)in_sizes; (void)n_in; (void)out_size; (void)ws_size;
  const float* X   = (const float*)d_in[0];
  const float* WQw = (const float*)d_in[1];
  const float* WQb = (const float*)d_in[2];
  const float* WKw = (const float*)d_in[3];
  const float* WKb = (const float*)d_in[4];
  const float* WVw = (const float*)d_in[5];
  const float* WVb = (const float*)d_in[6];
  const float* WOw = (const float*)d_in[7];
  const float* WOb = (const float*)d_in[8];
  const float* g1  = (const float*)d_in[9];
  const float* be1 = (const float*)d_in[10];
  const float* W1  = (const float*)d_in[11];
  const float* b1  = (const float*)d_in[12];
  const float* W2  = (const float*)d_in[13];
  const float* b2  = (const float*)d_in[14];
  const float* g2  = (const float*)d_in[15];
  const float* be2 = (const float*)d_in[16];

  char* ws = (char*)d_ws;
  const size_t MB = 1ull << 20;
  // Region 0..64MB: Xh(16)+Qh(16)+Kh(16)+Vth(16); reused as Ctx (over Xh) and
  // later as the 64MB MLP hidden buffer once attention is done.
  _Float16* Xh    = (_Float16*)(ws + 0);
  _Float16* Qh    = (_Float16*)(ws + 16 * MB);
  _Float16* Kh    = (_Float16*)(ws + 32 * MB);
  _Float16* Vth   = (_Float16*)(ws + 48 * MB);
  _Float16* Ctxh  = Xh;                       // Xh dead after QKV GEMM
  _Float16* Hh    = (_Float16*)(ws + 0);      // 64MB, Q/K/V dead after attn
  _Float16* Wqkvh = (_Float16*)(ws + 64 * MB);
  _Float16* WOh   = (_Float16*)(ws + 70 * MB);
  _Float16* W1h   = (_Float16*)(ws + 72 * MB);
  _Float16* W2h   = (_Float16*)(ws + 80 * MB);
  float*    Ybuf  = (float*)   (ws + 88 * MB);  // 32MB, reused for MLP2 out
  float*    X1    = (float*)   (ws + 120 * MB); // 32MB
  _Float16* X1h   = (_Float16*)(ws + 152 * MB); // 16MB

  const dim3 blk(256);
  const dim3 cvtg(1024);
  // fp32 -> f16 conversions (weights + activations)
  cvt_f16_kernel<<<cvtg, blk, 0, stream>>>(X,   Xh,                    M_ * DM_);
  cvt_f16_kernel<<<cvtg, blk, 0, stream>>>(WQw, Wqkvh,                 DM_ * DM_);
  cvt_f16_kernel<<<cvtg, blk, 0, stream>>>(WKw, Wqkvh + DM_ * DM_,     DM_ * DM_);
  cvt_f16_kernel<<<cvtg, blk, 0, stream>>>(WVw, Wqkvh + 2 * DM_ * DM_, DM_ * DM_);
  cvt_f16_kernel<<<cvtg, blk, 0, stream>>>(WOw, WOh,                   DM_ * DM_);
  cvt_f16_kernel<<<cvtg, blk, 0, stream>>>(W1,  W1h,                   DFF_ * DM_);
  cvt_f16_kernel<<<cvtg, blk, 0, stream>>>(W2,  W2h,                   DM_ * DFF_);

  // Fused QKV projection (N = 3072)
  qkv_gemm_kernel<<<dim3(3 * DM_ / 256, M_ / 128), blk, 0, stream>>>(
      Xh, Wqkvh, WQb, WKb, WVb, Qh, Kh, Vth);

  // Flash attention
  attn_kernel<<<dim3(S_ / 128, B_ * H_), blk, 0, stream>>>(Qh, Kh, Vth, Ctxh);

  // Output projection -> Ybuf (f32)
  gemm_wmma_kernel<0><<<dim3(DM_ / 256, M_ / 128), blk, 0, stream>>>(
      Ctxh, WOh, WOb, Ybuf, nullptr, M_, DM_, DM_);

  // LN1(y + X) -> X1 (f32) and X1h (f16)
  ln_kernel<<<dim3(M_), blk, 0, stream>>>(Ybuf, X, g1, be1, X1, X1h);

  // MLP hidden: relu(X1 @ W1^T + b1) -> Hh (f16)
  gemm_wmma_kernel<1><<<dim3(DFF_ / 256, M_ / 128), blk, 0, stream>>>(
      X1h, W1h, b1, nullptr, Hh, M_, DFF_, DM_);

  // MLP out: Hh @ W2^T + b2 -> Ybuf (f32)
  gemm_wmma_kernel<0><<<dim3(DM_ / 256, M_ / 128), blk, 0, stream>>>(
      Hh, W2h, b2, Ybuf, nullptr, M_, DM_, DFF_);

  // LN2(m + X1) -> d_out (f32)
  ln_kernel<<<dim3(M_), blk, 0, stream>>>(Ybuf, X1, g2, be2, (float*)d_out, nullptr);
}
